// MaskLockedLinear_25984552140896
// MI455X (gfx1250) — matrix operand order repaired and checked by reading, not compiled, not run on hardware
//
#include <hip/hip_runtime.h>
#include <math.h>

typedef unsigned short u16;
typedef unsigned int   u32;
typedef __attribute__((ext_vector_type(16))) __bf16 v16bf;
typedef __attribute__((ext_vector_type(8)))  float  v8f;

#define M_DIM 8192
#define K_DIM 4096
#define N_DIM 4096
#define NELEM_W (N_DIM * K_DIM)   // 16777216
#define NELEM_X (M_DIM * K_DIM)   // 33554432

// workspace layout (bytes)
#define WS_SCALARS_OFF 0u          // [0]=scale, [1]=threshold
#define WS_PART_OFF    256u        // 1024 blocks * 2 floats = 8 KB
#define WS_TW_OFF      65536u      // ternary bf16 weight, 32 MB
#define WS_XB_OFF      (65536u + 33554432u)  // bf16 x, 64 MB

#define RED_BLOCKS 1024

__device__ __forceinline__ u32 f2bf_rne(float f) {
    u32 u = __float_as_uint(f);
    u += 0x7FFFu + ((u >> 16) & 1u);
    return u >> 16;
}

// ---------- pass 1: per-block partial sums of |w*m| and (w*m)^2 ----------
__global__ void k_reduce1(const float* __restrict__ w, const float* __restrict__ m,
                          float* __restrict__ part, int n) {
    __shared__ float s1[256];
    __shared__ float s2[256];
    float a = 0.f, b = 0.f;
    for (int i = blockIdx.x * 256 + threadIdx.x; i < n; i += gridDim.x * 256) {
        float v = w[i] * m[i];
        a += fabsf(v);
        b += v * v;
    }
    s1[threadIdx.x] = a;
    s2[threadIdx.x] = b;
    __syncthreads();
    for (int s = 128; s > 0; s >>= 1) {
        if (threadIdx.x < s) {
            s1[threadIdx.x] += s1[threadIdx.x + s];
            s2[threadIdx.x] += s2[threadIdx.x + s];
        }
        __syncthreads();
    }
    if (threadIdx.x == 0) {
        part[2 * blockIdx.x + 0] = s1[0];
        part[2 * blockIdx.x + 1] = s2[0];
    }
}

// ---------- pass 2: finish reduction, compute scale & threshold ----------
__global__ void k_reduce2(const float* __restrict__ part, float* __restrict__ scalars,
                          int nblocks) {
    __shared__ float s1[256];
    __shared__ float s2[256];
    float a = 0.f, b = 0.f;
    for (int i = threadIdx.x; i < nblocks; i += 256) {
        a += part[2 * i + 0];
        b += part[2 * i + 1];
    }
    s1[threadIdx.x] = a;
    s2[threadIdx.x] = b;
    __syncthreads();
    for (int s = 128; s > 0; s >>= 1) {
        if (threadIdx.x < s) {
            s1[threadIdx.x] += s1[threadIdx.x + s];
            s2[threadIdx.x] += s2[threadIdx.x + s];
        }
        __syncthreads();
    }
    if (threadIdx.x == 0) {
        double S1 = (double)s1[0];
        double S2 = (double)s2[0];
        double N  = (double)NELEM_W;
        double mean = S1 / N;
        double var  = (S2 - S1 * S1 / N) / (N - 1.0);
        if (var < 0.0) var = 0.0;
        scalars[0] = (float)(mean + 1e-8);
        scalars[1] = (float)(0.5 * sqrt(var));
    }
}

// ---------- ternarize masked weight into bf16 {-1,0,+1} (branchless) ----------
__global__ void k_ternarize(const float* __restrict__ w, const float* __restrict__ m,
                            const float* __restrict__ scalars, u16* __restrict__ tw, int n) {
    const float thr = scalars[1];
    for (int i = blockIdx.x * 256 + threadIdx.x; i < n; i += gridDim.x * 256) {
        float v = w[i] * m[i];
        u16 o = (v > thr) ? (u16)0x3F80u : ((v < -thr) ? (u16)0xBF80u : (u16)0u);
        tw[i] = o;
    }
}

// ---------- convert x (f32) to bf16, vectorized ----------
__global__ void k_xconv(const float4* __restrict__ x4, uint2* __restrict__ xb, int n4) {
    for (int i = blockIdx.x * 256 + threadIdx.x; i < n4; i += gridDim.x * 256) {
        float4 f = x4[i];
        uint2 p;
        p.x = f2bf_rne(f.x) | (f2bf_rne(f.y) << 16);
        p.y = f2bf_rne(f.z) | (f2bf_rne(f.w) << 16);
        xb[i] = p;
    }
}

// ---------- WMMA GEMM: out[m,n] = scale * sum_k xb[m,k]*tw[n,k] + bias[n] ----------
// Block tile 128x128, 8 waves (4 M-stripes x 2 N-stripes), wave tile 32x64 = 2x4
// accumulators of 16x16. K stepped by 64 per LDS tile (two 16x16x32 WMMA chunks).
// Global -> LDS staging uses CDNA5 async DMA (global_load_async_to_lds_b128,
// ASYNCcnt), so no staging VGPRs and one barrier per K step.
#define KT         64
#define LDS_STRIDE 72   // 64 cols + 8 pad (144B row stride)

union FragU {
    uint4 u[2];
    v16bf v;
};

// copy 64 contiguous bytes global->LDS per lane, asynchronously.
// NOTE: the 24-bit instruction offset is added to BOTH the LDS and the global
// address (ISA 10.x async pseudocode), so one address pair covers 4 chunks.
__device__ __forceinline__ void async_copy64(u32 lds_addr, const u16* base, u32 voff) {
    asm volatile(
        "global_load_async_to_lds_b128 %0, %1, %2\n\t"
        "global_load_async_to_lds_b128 %0, %1, %2 offset:16\n\t"
        "global_load_async_to_lds_b128 %0, %1, %2 offset:32\n\t"
        "global_load_async_to_lds_b128 %0, %1, %2 offset:48"
        :
        : "v"(lds_addr), "v"(voff), "s"(base)
        : "memory");
}

__device__ __forceinline__ void wait_async0() {
    asm volatile("s_wait_asynccnt 0x0" ::: "memory");
}

__global__ __launch_bounds__(256) void k_gemm(const u16* __restrict__ xb,
                                              const u16* __restrict__ tw,
                                              const float* __restrict__ scalars,
                                              const float* __restrict__ bias,
                                              float* __restrict__ out) {
    __shared__ u16 ldsX[2][128][LDS_STRIDE];
    __shared__ u16 ldsW[2][128][LDS_STRIDE];

    const int tid  = threadIdx.x;
    const int lane = tid & 31;
    const int wave = tid >> 5;
    const int waveRow = wave >> 1;   // 0..3 -> 32-row stripes
    const int waveCol = wave & 1;    // 0..1 -> 64-col stripes
    const int lh = lane >> 4;        // lane half
    const int mr = lane & 15;

    const int row0 = blockIdx.y * 128;
    const int col0 = blockIdx.x * 128;

    // staging coords: thread t DMAs 32 bf16 of row (t>>1), col half (t&1)*32
    const int lr = tid >> 1;
    const int lc = (tid & 1) << 5;

    const u32 xoff = (u32)(((row0 + lr) * K_DIM + lc) * 2);  // byte offset into xb
    const u32 woff = (u32)(((col0 + lr) * K_DIM + lc) * 2);  // byte offset into tw

    // LDS byte offsets (low 32 bits of the LDS-aperture flat address)
    u32 ldsx[2], ldsw[2];
    ldsx[0] = (u32)(size_t)&ldsX[0][lr][lc];
    ldsx[1] = (u32)(size_t)&ldsX[1][lr][lc];
    ldsw[0] = (u32)(size_t)&ldsW[0][lr][lc];
    ldsw[1] = (u32)(size_t)&ldsW[1][lr][lc];

    const v8f vzero = {0.f, 0.f, 0.f, 0.f, 0.f, 0.f, 0.f, 0.f};
    v8f acc[2][4];
#pragma unroll
    for (int mt = 0; mt < 2; ++mt)
#pragma unroll
        for (int nt = 0; nt < 4; ++nt) acc[mt][nt] = vzero;

    // prologue: DMA K-tile 0 into buffer 0
    async_copy64(ldsx[0], xb, xoff);
    async_copy64(ldsw[0], tw, woff);
    wait_async0();
    __syncthreads();

    int buf = 0;
#pragma unroll 1
    for (int it = 0; it < K_DIM / KT; ++it) {
        const bool has_next = (it + 1) < (K_DIM / KT);
        if (has_next) {
            const u32 kb = (u32)((it + 1) * (KT * 2));  // byte advance along K
            async_copy64(ldsx[buf ^ 1], xb, xoff + kb);
            async_copy64(ldsw[buf ^ 1], tw, woff + kb);
        }

        const u16* Xs = (const u16*)&ldsX[buf][0][0];
        const u16* Ws = (const u16*)&ldsW[buf][0][0];

#pragma unroll
        for (int kc = 0; kc < 2; ++kc) {
            const int kcb = 32 * kc;
            // A frags: 16x32 bf16; lane holds M=mr, K = {8*lh..+7, 16+8*lh..+7}
            FragU a[2];
#pragma unroll
            for (int mt = 0; mt < 2; ++mt) {
                const int r = waveRow * 32 + mt * 16 + mr;
                a[mt].u[0] = *(const uint4*)(Xs + r * LDS_STRIDE + kcb + 8 * lh);
                a[mt].u[1] = *(const uint4*)(Xs + r * LDS_STRIDE + kcb + 16 + 8 * lh);
            }
            // B frags: 32x16 bf16; lane holds N=mr, K = {8*lh..+7, 16+8*lh..+7}
            FragU b[4];
#pragma unroll
            for (int nt = 0; nt < 4; ++nt) {
                const int r = waveCol * 64 + nt * 16 + mr;
                b[nt].u[0] = *(const uint4*)(Ws + r * LDS_STRIDE + kcb + 8 * lh);
                b[nt].u[1] = *(const uint4*)(Ws + r * LDS_STRIDE + kcb + 16 + 8 * lh);
            }
#pragma unroll
            for (int mt = 0; mt < 2; ++mt)
#pragma unroll
                for (int nt = 0; nt < 4; ++nt)
                    acc[mt][nt] = __builtin_amdgcn_wmma_f32_16x16x32_bf16(
                        false, a[mt].v, false, b[nt].v, (short)0, acc[mt][nt], false, false);
        }

        if (has_next) wait_async0();  // this wave's DMAs into lds[buf^1] are done
        __syncthreads();              // all waves: reads of lds[buf] and DMAs done
        buf ^= 1;
    }

    const float scale = scalars[0];
    // C/D layout: VGPR r -> M = base + 8*lh + r ; N = base + mr
#pragma unroll
    for (int nt = 0; nt < 4; ++nt) {
        const int col = col0 + waveCol * 64 + nt * 16 + mr;
        const float bv = bias[col];
#pragma unroll
        for (int mt = 0; mt < 2; ++mt) {
            const int rowb = row0 + waveRow * 32 + mt * 16 + 8 * lh;
#pragma unroll
            for (int r = 0; r < 8; ++r) {
                out[(size_t)(rowb + r) * N_DIM + col] = scale * acc[mt][nt][r] + bv;
            }
        }
    }
}

extern "C" void kernel_launch(void* const* d_in, const int* in_sizes, int n_in,
                              void* d_out, int out_size, void* d_ws, size_t ws_size,
                              hipStream_t stream) {
    const float* x    = (const float*)d_in[0];
    const float* w    = (const float*)d_in[1];
    const float* msk  = (const float*)d_in[2];
    const float* bias = (const float*)d_in[3];
    float* out = (float*)d_out;

    char* ws = (char*)d_ws;
    float* scalars = (float*)(ws + WS_SCALARS_OFF);
    float* part    = (float*)(ws + WS_PART_OFF);
    u16*   twb     = (u16*)(ws + WS_TW_OFF);
    u16*   xbb     = (u16*)(ws + WS_XB_OFF);

    k_reduce1<<<RED_BLOCKS, 256, 0, stream>>>(w, msk, part, NELEM_W);
    k_reduce2<<<1, 256, 0, stream>>>(part, scalars, RED_BLOCKS);
    k_ternarize<<<4096, 256, 0, stream>>>(w, msk, scalars, twb, NELEM_W);
    k_xconv<<<4096, 256, 0, stream>>>((const float4*)x, (uint2*)xbb, NELEM_X / 4);

    dim3 grid(N_DIM / 128, M_DIM / 128);  // 32 x 64
    k_gemm<<<grid, 256, 0, stream>>>(xbb, twb, scalars, bias, out);
}
